// CustomActivatedLSTMCell_5712306503741
// MI455X (gfx1250) — compile-verified, matrix operand order checked by reading, not run on hardware
//
#include <hip/hip_runtime.h>

typedef __bf16 v16bf __attribute__((ext_vector_type(16)));
typedef float  v8f   __attribute__((ext_vector_type(8)));
typedef unsigned int v4u __attribute__((ext_vector_type(4)));
typedef int v8i __attribute__((ext_vector_type(8)));
typedef int v4i __attribute__((ext_vector_type(4)));

#define B_DIM  4096
#define IN_DIM 1024
#define H_DIM  2048

// ---- TDM-pipelined GEMM kernel geometry ----
#define TM  128      // batch rows per workgroup
#define TN  32       // j-columns per workgroup (x4 gate chunks)
#define KT  64       // K elements staged per pipeline stage (2 WMMA k-steps)
#define LDSTR 72     // LDS row stride in bf16 elems (144 B: 16B pad per 128B row)
#define NST (IN_DIM / KT + H_DIM / KT)   // 16 + 32 = 48 stages
#define TILE_ELEMS (128 * LDSTR)         // per A- or W-buffer (rows x stride)

#if __has_builtin(__builtin_amdgcn_tensor_load_to_lds)
#define USE_TDM 1
#else
#define USE_TDM 0
#endif

union FragU { v16bf bf; uint4 q[2]; };

// ---------------- fp32 -> bf16 convert pre-pass (bandwidth bound) -------------
__global__ __launch_bounds__(256)
void cvt_bf16_kernel(const float* __restrict__ src, uint2* __restrict__ dst, int n4) {
  int i = blockIdx.x * 256 + threadIdx.x;
  if (i < n4) {
    const float4 v = ((const float4*)src)[i];
    union { __bf16 b[4]; uint2 u; } o;
    o.b[0] = (__bf16)v.x; o.b[1] = (__bf16)v.y;
    o.b[2] = (__bf16)v.z; o.b[3] = (__bf16)v.w;
    dst[i] = o.u;
  }
}

// ---------------- TDM issue helper (D# per CDNA5 ISA 8.3-8.6) -----------------
__device__ __forceinline__ void tdm_issue(unsigned lds_off, const void* gptr,
                                          unsigned tile_d0, unsigned tile_d1,
                                          unsigned tile_d2,
                                          unsigned long long stride0,
                                          unsigned long long stride1) {
#if USE_TDM
  unsigned long long ga = (unsigned long long)gptr;
  // group0: count=1 | lds_addr | global_addr[56:0] | type=2
  v4u g0 = { 1u, lds_off, (unsigned)ga,
             (unsigned)((ga >> 32) & 0x01FFFFFFu) | 0x80000000u };
  const unsigned td0 = 1u << 30, td1 = 1u << 30;   // generous OOB dims (all in-bounds)
  v8i g1;
  // data_size=1(2B), pad_enable, pad_interval=4 (every 128B), pad_amount=3 (16B)
  g1[0] = (1 << 16) | (1 << 20) | (4 << 22) | (3 << 25);
  g1[1] = (int)((td0 & 0xFFFFu) << 16);                               // td0[15:0]
  g1[2] = (int)(((td0 >> 16) & 0xFFFFu) | ((td1 & 0xFFFFu) << 16));   // td0[31:16]|td1[15:0]
  g1[3] = (int)(((td1 >> 16) & 0xFFFFu) | (tile_d0 << 16));           // td1[31:16]|tile0
  g1[4] = (int)(tile_d1 | (tile_d2 << 16));                           // tile1|tile2
  g1[5] = (int)(stride0 & 0xFFFFFFFFu);                               // dim0_stride lo
  g1[6] = (int)(((stride0 >> 32) & 0xFFFFu) | ((stride1 & 0xFFFFu) << 16));
  g1[7] = (int)((stride1 >> 16) & 0xFFFFFFFFu);                       // dim1_stride hi
  v4i g2 = { 8, 0, 0, 0 };   // tensor_dim2 >= tile_dim2; dim2_stride/tile3 unused
  v4i g3 = { 0, 0, 0, 0 };
  v8i g4 = { 0, 0, 0, 0, 0, 0, 0, 0 };   // extra group slot (clang-23 6-arg form)
  __builtin_amdgcn_tensor_load_to_lds(g0, g1, g2, g3, g4, 0);
#else
  (void)lds_off; (void)gptr; (void)tile_d0; (void)tile_d1; (void)tile_d2;
  (void)stride0; (void)stride1;
#endif
}

__device__ __forceinline__ void wait_tensor0() {
#if __has_builtin(__builtin_amdgcn_s_wait_tensorcnt)
  __builtin_amdgcn_s_wait_tensorcnt(0);
#else
  asm volatile("s_wait_tensorcnt 0x0" ::: "memory");
#endif
}

// stage s -> (source, K, kk)
__device__ __forceinline__ void stage_src(int s, const __bf16* xb, const __bf16* hb,
                                          const __bf16* wihb, const __bf16* whhb,
                                          const __bf16** A, const __bf16** W,
                                          int* K, int* kk) {
  if (s < IN_DIM / KT) { *A = xb; *W = wihb; *K = IN_DIM; *kk = s * KT; }
  else                 { *A = hb; *W = whhb; *K = H_DIM;  *kk = (s - IN_DIM / KT) * KT; }
}

// 16 WMMAs per stage: 4 M-blocks x 2 N-blocks x 2 k-substeps
__device__ __forceinline__ void compute_stage(const __bf16* bA, const __bf16* bW,
                                              v8f (&acc)[4][2],
                                              int g, int mh, int lrow, int kb) {
  #pragma unroll
  for (int ks = 0; ks < KT; ks += 32) {
    FragU fb[2];
    #pragma unroll
    for (int nb = 0; nb < 2; ++nb) {
      const __bf16* bp = bW + (g * 32 + nb * 16 + lrow) * LDSTR + ks + kb;
      fb[nb].q[0] = *(const uint4*)(bp);
      fb[nb].q[1] = *(const uint4*)(bp + 16);
    }
    #pragma unroll
    for (int mb = 0; mb < 4; ++mb) {
      FragU fa;
      const __bf16* ap = bA + (mh * 64 + mb * 16 + lrow) * LDSTR + ks + kb;
      fa.q[0] = *(const uint4*)(ap);
      fa.q[1] = *(const uint4*)(ap + 16);
      #pragma unroll
      for (int nb = 0; nb < 2; ++nb)
        acc[mb][nb] = __builtin_amdgcn_wmma_f32_16x16x32_bf16(
            false, fa.bf, false, fb[nb].bf, (short)0, acc[mb][nb], false, false);
    }
  }
}

// ---------------- main GEMM + fused LSTM kernel (bf16 inputs) -----------------
__global__ __launch_bounds__(256)
void lstm_wmma_tdm_kernel(const __bf16* __restrict__ xb, const __bf16* __restrict__ hb,
                          const float* __restrict__ c,
                          const __bf16* __restrict__ wihb, const __bf16* __restrict__ whhb,
                          const float* __restrict__ b_ih, const float* __restrict__ b_hh,
                          float* __restrict__ out)
{
  __shared__ __bf16 sA[2 * TILE_ELEMS];    // double-buffered activation tiles
  __shared__ __bf16 sW[2 * TILE_ELEMS];    // double-buffered 4-gate weight tiles
  __shared__ float  sG[4 * TM * TN];       // 64 KB fp32 gate staging

  const int t     = threadIdx.x;
  const int lane  = t & 31;
  const int wave  = t >> 5;
  const int g     = wave >> 1;    // gate chunk 0..3
  const int mh    = wave & 1;     // which 64-row half of the 128-row tile
  const int lrow  = lane & 15;
  const int lhalf = lane >> 4;
  const int kb    = lhalf * 8;    // ISA bf16 A/B fragment k-split

  const int m0 = blockIdx.y * TM;
  const int n0 = blockIdx.x * TN;

  v8f acc[4][2] = {};

#if USE_TDM
  const unsigned ldsA0 = (unsigned)(unsigned long long)(void*)&sA[0];
  const unsigned ldsW0 = (unsigned)(unsigned long long)(void*)&sW[0];
  if (t < 32) {   // wave 0 is the DMA engine driver (EXEC ignored by tensor ops)
    const __bf16 *A, *W; int K, kk;
    stage_src(0, xb, hb, wihb, whhb, &A, &W, &K, &kk);
    tdm_issue(ldsA0, A + (size_t)m0 * K + kk, KT, TM, 0, K, 0);
    tdm_issue(ldsW0, W + (size_t)n0 * K + kk, KT, TN, 4, K,
              (unsigned long long)H_DIM * K);
  }
#endif

  for (int s = 0; s < NST; ++s) {
#if USE_TDM
    const int cur = s & 1;
    if (t < 32) wait_tensor0();          // stage s tiles landed in LDS
    __syncthreads();                     // publish to all waves; frees buf cur^1
    if (t < 32 && (s + 1) < NST) {       // overlap: DMA stage s+1 during compute s
      const __bf16 *A, *W; int K, kk;
      stage_src(s + 1, xb, hb, wihb, whhb, &A, &W, &K, &kk);
      tdm_issue(ldsA0 + (cur ^ 1) * (TILE_ELEMS * 2),
                A + (size_t)m0 * K + kk, KT, TM, 0, K, 0);
      tdm_issue(ldsW0 + (cur ^ 1) * (TILE_ELEMS * 2),
                W + (size_t)n0 * K + kk, KT, TN, 4, K,
                (unsigned long long)H_DIM * K);
    }
    const __bf16* bA = sA + cur * TILE_ELEMS;
    const __bf16* bW = sW + cur * TILE_ELEMS;
#else
    // synchronous fallback: stage bf16 tiles via VGPRs (no conversion needed)
    const __bf16 *A, *W; int K, kk;
    stage_src(s, xb, hb, wihb, whhb, &A, &W, &K, &kk);
    __syncthreads();
    #pragma unroll
    for (int r = 0; r < 4; ++r) {        // A tile: 128 rows x 8 16B-chunks
      int id = t + r * 256, row = id >> 3, ch = id & 7;
      *(uint4*)(sA + row * LDSTR + ch * 8) =
          *(const uint4*)(A + (size_t)(m0 + row) * K + kk + ch * 8);
    }
    #pragma unroll
    for (int r = 0; r < 4; ++r) {        // W tiles: 4 gates x 32 rows x 8 chunks
      int id = t + r * 256, gg = id >> 8, rw = (id >> 3) & 31, ch = id & 7;
      *(uint4*)(sW + (gg * 32 + rw) * LDSTR + ch * 8) =
          *(const uint4*)(W + (size_t)(gg * H_DIM + n0 + rw) * K + kk + ch * 8);
    }
    __syncthreads();
    const __bf16* bA = sA;
    const __bf16* bW = sW;
#endif
    compute_stage(bA, bW, acc, g, mh, lrow, kb);
  }

  // ---- spill accumulators (C layout: lanes 0-15 M=r, 16-31 M=r+8) ----
  __syncthreads();
  #pragma unroll
  for (int mb = 0; mb < 4; ++mb)
    #pragma unroll
    for (int nb = 0; nb < 2; ++nb) {
      int mbase = mh * 64 + mb * 16 + lhalf * 8;
      #pragma unroll
      for (int r = 0; r < 8; ++r)
        sG[g * (TM * TN) + (mbase + r) * TN + nb * 16 + lrow] = acc[mb][nb][r];
    }
  __syncthreads();

  // ---- fused LSTM cell update (fp32) ----
  for (int e = t; e < TM * TN; e += 256) {
    int m = e >> 5, n = e & 31;
    int row = m0 + m, j = n0 + n;
    float fg = sG[0 * TM * TN + e] + b_ih[0 * H_DIM + j] + b_hh[0 * H_DIM + j];
    float ig = sG[1 * TM * TN + e] + b_ih[1 * H_DIM + j] + b_hh[1 * H_DIM + j];
    float cg = sG[2 * TM * TN + e] + b_ih[2 * H_DIM + j] + b_hh[2 * H_DIM + j];
    float og = sG[3 * TM * TN + e] + b_ih[3 * H_DIM + j] + b_hh[3 * H_DIM + j];
    float ft = 1.0f / (1.0f + __expf(-fg));
    float it = 1.0f / (1.0f + __expf(-ig));
    float ot = 1.0f / (1.0f + __expf(-og));
    float ct = c[(size_t)row * H_DIM + j] * ft + __sinf(cg) * it;
    float ht = ot * __sinf(ct);
    out[(size_t)row * H_DIM + j] = ht;
    out[(size_t)B_DIM * H_DIM + (size_t)row * H_DIM + j] = ct;
  }
}

// ================= round-1 fused-conversion kernel (ws fallback) ==============
__device__ __forceinline__ unsigned int bf16pack(float a, float b) {
  union { float f; unsigned int u; } x, y;
  x.f = a; y.f = b;
  unsigned int ua = x.u + (0x7FFFu + ((x.u >> 16) & 1u));
  unsigned int ub = y.u + (0x7FFFu + ((y.u >> 16) & 1u));
  return (ua >> 16) | (ub & 0xFFFF0000u);
}

#define FTM 64
#define FSTR 40

__global__ __launch_bounds__(256)
void lstm_wmma_fused(const float* __restrict__ x,    const float* __restrict__ h,
                     const float* __restrict__ c,    const float* __restrict__ w_ih,
                     const float* __restrict__ w_hh, const float* __restrict__ b_ih,
                     const float* __restrict__ b_hh, float* __restrict__ out)
{
  __shared__ __bf16 sA[FTM * FSTR];
  __shared__ __bf16 sW[4 * TN * FSTR];
  __shared__ float  sG[4 * FTM * TN];

  const int t = threadIdx.x, lane = t & 31, wave = t >> 5;
  const int g = wave >> 1, nb = wave & 1;
  const int lrow = lane & 15, lhalf = lane >> 4, kb = lhalf * 8;
  const int m0 = blockIdx.y * FTM, n0 = blockIdx.x * TN;

  v8f acc[4] = {};
  for (int phase = 0; phase < 2; ++phase) {
    const float* As = phase ? h : x;
    const float* Ws = phase ? w_hh : w_ih;
    const int K = phase ? H_DIM : IN_DIM;
    for (int kk = 0; kk < K; kk += 32) {
      __syncthreads();
      #pragma unroll
      for (int r = 0; r < 2; ++r) {
        int id = t + r * 256, row = id >> 3, col = (id & 7) * 4;
        const float4 v = *(const float4*)(As + (size_t)(m0 + row) * K + kk + col);
        *(uint2*)(&sA[row * FSTR + col]) = make_uint2(bf16pack(v.x, v.y), bf16pack(v.z, v.w));
      }
      #pragma unroll
      for (int r = 0; r < 4; ++r) {
        int id = t + r * 256, gg = id >> 8, rw = (id >> 3) & 31, col = (id & 7) * 4;
        const float4 v = *(const float4*)(Ws + (size_t)(gg * H_DIM + n0 + rw) * K + kk + col);
        *(uint2*)(&sW[(gg * TN + rw) * FSTR + col]) = make_uint2(bf16pack(v.x, v.y), bf16pack(v.z, v.w));
      }
      __syncthreads();
      FragU fb;
      const __bf16* bp = &sW[(g * TN + nb * 16 + lrow) * FSTR + kb];
      fb.q[0] = *(const uint4*)(bp); fb.q[1] = *(const uint4*)(bp + 16);
      #pragma unroll
      for (int mb = 0; mb < 4; ++mb) {
        FragU fa;
        const __bf16* ap = &sA[(mb * 16 + lrow) * FSTR + kb];
        fa.q[0] = *(const uint4*)(ap); fa.q[1] = *(const uint4*)(ap + 16);
        acc[mb] = __builtin_amdgcn_wmma_f32_16x16x32_bf16(
            false, fa.bf, false, fb.bf, (short)0, acc[mb], false, false);
      }
    }
  }
  __syncthreads();
  #pragma unroll
  for (int mb = 0; mb < 4; ++mb) {
    int mbase = mb * 16 + lhalf * 8;
    #pragma unroll
    for (int r = 0; r < 8; ++r)
      sG[g * (FTM * TN) + (mbase + r) * TN + nb * 16 + lrow] = acc[mb][r];
  }
  __syncthreads();
  for (int e = t; e < FTM * TN; e += 256) {
    int m = e >> 5, n = e & 31;
    int row = m0 + m, j = n0 + n;
    float fg = sG[0 * FTM * TN + e] + b_ih[0 * H_DIM + j] + b_hh[0 * H_DIM + j];
    float ig = sG[1 * FTM * TN + e] + b_ih[1 * H_DIM + j] + b_hh[1 * H_DIM + j];
    float cg = sG[2 * FTM * TN + e] + b_ih[2 * H_DIM + j] + b_hh[2 * H_DIM + j];
    float og = sG[3 * FTM * TN + e] + b_ih[3 * H_DIM + j] + b_hh[3 * H_DIM + j];
    float ft = 1.0f / (1.0f + __expf(-fg));
    float it = 1.0f / (1.0f + __expf(-ig));
    float ot = 1.0f / (1.0f + __expf(-og));
    float ct = c[(size_t)row * H_DIM + j] * ft + __sinf(cg) * it;
    float ht = ot * __sinf(ct);
    out[(size_t)row * H_DIM + j] = ht;
    out[(size_t)B_DIM * H_DIM + (size_t)row * H_DIM + j] = ct;
  }
}

// =============================== launcher =====================================
extern "C" void kernel_launch(void* const* d_in, const int* in_sizes, int n_in,
                              void* d_out, int out_size, void* d_ws, size_t ws_size,
                              hipStream_t stream) {
  const float* x    = (const float*)d_in[0];
  const float* h    = (const float*)d_in[1];
  const float* c    = (const float*)d_in[2];
  const float* w_ih = (const float*)d_in[3];
  const float* w_hh = (const float*)d_in[4];
  const float* b_ih = (const float*)d_in[5];
  const float* b_hh = (const float*)d_in[6];
  float* out = (float*)d_out;

  const size_t nx  = (size_t)B_DIM * IN_DIM;       // 4.19M
  const size_t nh  = (size_t)B_DIM * H_DIM;        // 8.39M
  const size_t nwi = (size_t)4 * H_DIM * IN_DIM;   // 8.39M
  const size_t nwh = (size_t)4 * H_DIM * H_DIM;    // 16.8M
  const size_t need = (nx + nh + nwi + nwh) * sizeof(__bf16);   // 72 MB

  if (ws_size >= need) {
    __bf16* xb   = (__bf16*)d_ws;
    __bf16* hb   = xb + nx;
    __bf16* wihb = hb + nh;
    __bf16* whhb = wihb + nwi;
    // fp32 -> bf16 pre-pass (one-shot per call, bandwidth bound)
    cvt_bf16_kernel<<<(int)(nx  / 4 / 256), 256, 0, stream>>>(x,    (uint2*)xb,   (int)(nx  / 4));
    cvt_bf16_kernel<<<(int)(nh  / 4 / 256), 256, 0, stream>>>(h,    (uint2*)hb,   (int)(nh  / 4));
    cvt_bf16_kernel<<<(int)(nwi / 4 / 256), 256, 0, stream>>>(w_ih, (uint2*)wihb, (int)(nwi / 4));
    cvt_bf16_kernel<<<(int)(nwh / 4 / 256), 256, 0, stream>>>(w_hh, (uint2*)whhb, (int)(nwh / 4));
    dim3 grid(H_DIM / TN, B_DIM / TM);   // 64 x 32
    lstm_wmma_tdm_kernel<<<grid, 256, 0, stream>>>(xb, hb, c, wihb, whhb, b_ih, b_hh, out);
  } else {
    dim3 grid(H_DIM / TN, B_DIM / FTM);  // 64 x 64
    lstm_wmma_fused<<<grid, 256, 0, stream>>>(x, h, c, w_ih, w_hh, b_ih, b_hh, out);
  }
}
